// TopologyAwareLoss_62577673502906
// MI455X (gfx1250) — compile-verified, matrix operand order checked by reading, not run on hardware
//
#include <hip/hip_runtime.h>
#include <hip/hip_bf16.h>

// ---------------------------------------------------------------------------
// TopologyAwareLoss for MI455X (gfx1250, wave32).
//   pred   : f32 [4,19,512,512]   target : i32 [4,512,512]   out : f32 [1]
// Pipeline:
//   1) softmax over C (HBM-bound, fp32, prefetched strided channel reads)
//   2) fused Laplacian-boundary + count-based BCE (WMMA f16 wave reduction)
//   3) connected components: init -> 6x(LDS-tile min-sweeps + pointer jump)
//   4) per-plane component counts (WMMA wave reduction), finalize scalar loss
// Workspace: 80MB soft + 159MB labels + 624B counters (~239MB total).
// ---------------------------------------------------------------------------

typedef __attribute__((ext_vector_type(16))) _Float16 v16h;
typedef __attribute__((ext_vector_type(8)))  float    v8f;

namespace {
constexpr int BB = 4, CC = 19, HH = 512, WW = 512;
constexpr int HW = HH * WW;           // 262144
constexpr int BC = BB * CC;           // 76
constexpr int NPLANES = 2 * BC;       // 152
constexpr int BIGL = NPLANES * HW;    // 39845888 (sentinel & label array size)
constexpr size_t SOFT_BYTES  = (size_t)BC * HW * 4;       // 79,691,776
constexpr size_t LABEL_BYTES = (size_t)BIGL * 4;          // 159,383,552
}

__device__ __forceinline__ int imin(int a, int b) { return a < b ? a : b; }

// Wave32 WMMA count reducer: every lane contributes up to 3 small integer
// counts (exact in f16); column n of D accumulates counter n across the wave.
// A row m is formed by lanes m (K=0..2 via elems 0..2) and m+16 (K=8..10).
// B selector: B[K=n][n]=1 (lane n, elem n) and B[K=n+8][n]=1 (lane n+16, elem n).
// Result: lanes {n, n+16} hold column n; sum of their 8 f32 regs = wave total.
__device__ __forceinline__ float wmma_wave_count3(int lane, float c0, float c1, float c2) {
  v16h a = {};
  a[0] = (_Float16)c0;
  a[1] = (_Float16)c1;
  a[2] = (_Float16)c2;
  v16h bm = {};
  if (lane == 0 || lane == 16) bm[0] = (_Float16)1.0f;
  if (lane == 1 || lane == 17) bm[1] = (_Float16)1.0f;
  if (lane == 2 || lane == 18) bm[2] = (_Float16)1.0f;
  v8f acc = {};
  acc = __builtin_amdgcn_wmma_f32_16x16x32_f16(
      /*neg_a=*/false, a, /*neg_b=*/false, bm,
      /*c_mod=*/(short)0, acc, /*reuse_a=*/false, /*reuse_b=*/false);
  return acc[0] + acc[1] + acc[2] + acc[3] + acc[4] + acc[5] + acc[6] + acc[7];
}

// ---------------------------------------------------------------- counters --
__global__ void zero_cnt_kernel(unsigned* __restrict__ cnt) {
  int t = threadIdx.x;
  if (t < 4 + NPLANES) cnt[t] = 0u;   // [0]=N_x1 [1]=N_x1y0 [2]=N_valid [3]=pad, comp[152]
}

// ----------------------------------------------------------------- softmax --
__global__ void softmax_kernel(const float* __restrict__ pred,
                               float* __restrict__ soft) {
  int idx = blockIdx.x * blockDim.x + threadIdx.x;   // over B*HW
  int b  = idx / HW;
  int hw = idx - b * HW;
  const float* p = pred + (size_t)b * CC * HW + hw;
  // prefetch the strided channel walk (global_prefetch_b8 path)
  #pragma unroll
  for (int c = 0; c < CC; c += 4) __builtin_prefetch(p + (size_t)c * HW, 0, 1);
  float v[CC];
  float m = -3.4e38f;
  #pragma unroll
  for (int c = 0; c < CC; ++c) { v[c] = p[(size_t)c * HW]; m = fmaxf(m, v[c]); }
  float s = 0.f;
  #pragma unroll
  for (int c = 0; c < CC; ++c) { v[c] = __expf(v[c] - m); s += v[c]; }
  float inv = 1.f / s;
  float* o = soft + (size_t)b * CC * HW + hw;
  #pragma unroll
  for (int c = 0; c < CC; ++c) o[(size_t)c * HW] = v[c] * inv;
}

// ---------------------------------------------- boundary + count-based BCE --
// 64x64 tile per block (256 thr, 16 px/thread), 66x66 halo staged in LDS.
// Binary-map BCE reduces to 3 integer counts; one WMMA reduces each wave.
__global__ void boundary_bce_kernel(const float* __restrict__ soft,
                                    const int* __restrict__ target,
                                    unsigned* __restrict__ cnt) {
  constexpr int HALO = 66;
  __shared__ float sf[HALO * HALO];
  __shared__ int   st[HALO * HALO];

  int plane = blockIdx.x >> 6;          // 0..75  (b*C + c)
  int tile  = blockIdx.x & 63;          // 8x8 tiles of 64x64
  int ty = tile >> 3, tx = tile & 7;
  int gy0 = ty * 64, gx0 = tx * 64;
  int b = plane / CC, c = plane - b * CC;
  int tid = threadIdx.x;

  for (int i = tid; i < HALO * HALO; i += 256) {
    int yy = i / HALO, xx = i - yy * HALO;
    int y = gy0 + yy - 1, x = gx0 + xx - 1;
    bool in = (y >= 0) & (y < HH) & (x >= 0) & (x < WW);
    sf[i] = in ? soft[(size_t)plane * HW + y * WW + x] : 0.f;
    st[i] = in ? target[(size_t)b * HW + y * WW + x] : 255;   // 255 => one_hot 0
  }
  __syncthreads();

  int cx = 0, cxy = 0, cv = 0;
  #pragma unroll 4
  for (int i = 0; i < 16; ++i) {
    int p  = tid + 256 * i;             // 0..4095 inside 64x64 tile
    int ly = p >> 6, lx = p & 63;
    int ci = (ly + 1) * HALO + (lx + 1);
    float ctr = sf[ci];
    float ns  = sf[ci - HALO - 1] + sf[ci - HALO] + sf[ci - HALO + 1]
              + sf[ci - 1]                         + sf[ci + 1]
              + sf[ci + HALO - 1] + sf[ci + HALO] + sf[ci + HALO + 1];
    float lapP = 8.f * ctr - ns;
    int tc = st[ci];
    bool valid = (tc != 255);
    int ohc = (tc == c);
    int ohs = (st[ci - HALO - 1] == c) + (st[ci - HALO] == c) + (st[ci - HALO + 1] == c)
            + (st[ci - 1] == c)                               + (st[ci + 1] == c)
            + (st[ci + HALO - 1] == c) + (st[ci + HALO] == c) + (st[ci + HALO + 1] == c);
    int lapT = 8 * ohc - ohs;           // |lapT|>0.1  <=>  lapT != 0
    bool xb = (fabsf(lapP) > 0.1f) && valid;
    bool yb = (lapT != 0) && valid;
    cx  += xb;
    cxy += (xb && !yb);
    cv  += valid;
  }

  int lane = threadIdx.x & 31;
  float s = wmma_wave_count3(lane, (float)cx, (float)cxy, (float)cv);
  unsigned u = (unsigned)(s + 0.5f);
  if (lane == 0 || lane == 16) atomicAdd(&cnt[0], u);
  if (lane == 1 || lane == 17) atomicAdd(&cnt[1], u);
  if ((lane == 2 || lane == 18) && c == 0) atomicAdd(&cnt[2], u);
}

// ------------------------------------------------------------ CC: init -----
__global__ void cc_init_kernel(const float* __restrict__ soft,
                               const int* __restrict__ target,
                               int* __restrict__ labels) {
  int idx = blockIdx.x * 256 + threadIdx.x;    // < BIGL
  int n = idx / HW;
  int r = idx - n * HW;
  bool mask;
  if (n < BC) {                                // pred masks
    int b = n / CC;
    int t = target[(size_t)b * HW + r];
    mask = (soft[idx] > 0.5f) && (t != 255);
  } else {                                     // target masks
    int n2 = n - BC;
    int b = n2 / CC, c = n2 - b * CC;
    mask = (target[(size_t)b * HW + r] == c);  // c<255 => implies valid
  }
  labels[idx] = mask ? idx : BIGL;
}

// ------------------------------------- CC: LDS tile min-label propagation --
// 32x32 tile + halo in LDS, 16 local 4-neighbor min sweeps per launch.
// Labels only decrease; each interior cell is owned by one block => plain store.
__global__ void cc_prop_kernel(int* __restrict__ labels) {
  constexpr int CHW = 34;
  __shared__ int sl[CHW * CHW];
  int plane = blockIdx.x >> 8;                 // 0..151
  int tile  = blockIdx.x & 255;                // 16x16 tiles of 32x32
  int ty = tile >> 4, tx = tile & 15;
  int gy0 = ty * 32, gx0 = tx * 32;
  int pbase = plane * HW;
  int tid = threadIdx.x;

  for (int i = tid; i < CHW * CHW; i += 256) {
    int yy = i / CHW, xx = i - yy * CHW;
    int y = gy0 + yy - 1, x = gx0 + xx - 1;
    bool in = (y >= 0) & (y < HH) & (x >= 0) & (x < WW);
    sl[i] = in ? labels[pbase + y * WW + x] : BIGL;
  }
  __syncthreads();

  for (int it = 0; it < 16; ++it) {
    int nv[4];
    #pragma unroll
    for (int j = 0; j < 4; ++j) {
      int p  = tid + 256 * j;                  // 0..1023
      int ly = p >> 5, lx = p & 31;
      int ci = (ly + 1) * CHW + (lx + 1);
      int l = sl[ci];
      int m = imin(imin(sl[ci - CHW], sl[ci + CHW]), imin(sl[ci - 1], sl[ci + 1]));
      nv[j] = (l == BIGL) ? BIGL : imin(l, m);
    }
    __syncthreads();
    #pragma unroll
    for (int j = 0; j < 4; ++j) {
      int p  = tid + 256 * j;
      int ly = p >> 5, lx = p & 31;
      sl[(ly + 1) * CHW + (lx + 1)] = nv[j];
    }
    __syncthreads();
  }

  #pragma unroll
  for (int j = 0; j < 4; ++j) {
    int p  = tid + 256 * j;
    int ly = p >> 5, lx = p & 31;
    labels[pbase + (gy0 + ly) * WW + (gx0 + lx)] = sl[(ly + 1) * CHW + (lx + 1)];
  }
}

// -------------------------------------- CC: pointer jumping (path chase) ---
__global__ void cc_jump_kernel(int* __restrict__ labels) {
  int idx = blockIdx.x * 256 + threadIdx.x;
  int l0 = labels[idx];
  if (l0 >= BIGL) return;
  int l = l0;
  for (int j = 0; j < 8; ++j) {
    int l2 = labels[l];                        // same component, l2 <= l
    if (l2 == l) break;
    l = l2;
  }
  if (l != l0) labels[idx] = l;
}

// ------------------------------------------------------------ CC: counting --
// Streams 160MB of labels; keep it bandwidth-bound: WMMA collapses each wave's
// counts (<=128, exact in f16) so only 2 LDS atomics/wave + 1 global/block.
__global__ void cc_count_kernel(const int* __restrict__ labels,
                                int* __restrict__ comp) {
  __shared__ int lc;
  int tid = threadIdx.x;
  if (tid == 0) lc = 0;
  __syncthreads();
  int base = blockIdx.x * 1024;                // 1024 | HW => single plane/block
  int plane = base / HW;
  int local = 0;
  #pragma unroll
  for (int j = 0; j < 4; ++j) {
    int i = base + tid + 256 * j;
    local += (labels[i] == i);
  }
  int lane = tid & 31;
  float s = wmma_wave_count3(lane, (float)local, 0.f, 0.f);  // full EXEC here
  if (lane == 0 || lane == 16) {
    int w = (int)(s + 0.5f);
    if (w) atomicAdd(&lc, w);
  }
  __syncthreads();
  if (tid == 0 && lc) atomicAdd(&comp[plane], lc);
}

// ---------------------------------------------------------------- finalize --
__global__ void finalize_kernel(const unsigned* __restrict__ cnt,
                                const int* __restrict__ comp,
                                float* __restrict__ out) {
  __shared__ float red[128];
  int t = threadIdx.x;
  float term = 0.f;
  if (t < BC) {
    float pc = (float)comp[t];
    float tc = (float)comp[t + BC];
    term = fabsf(pc - tc) * (comp[t + BC] > 0 ? 1.f : 0.f);
  }
  red[t] = term;
  __syncthreads();
  for (int s = 64; s > 0; s >>= 1) {
    if (t < s) red[t] += red[t + s];
    __syncthreads();
  }
  if (t == 0) {
    const float LN2 = 0.69314718056f;
    const float S1  = 0.31326168751f;          // log1p(exp(-1))
    float T = (float)BC * (float)HW;           // 19,922,944 (exact in f32)
    float nx1   = (float)cnt[0];
    float nx1y0 = (float)cnt[1];
    float nv    = (float)cnt[2];
    float boundary = (T * LN2 + nx1 * (S1 - LN2) + nx1y0) / (nv + 1e-8f);
    float conn = red[0] / ((float)BC + 1e-8f);
    out[0] = boundary + 0.1f * conn;
  }
}

// ---------------------------------------------------------------------------
extern "C" void kernel_launch(void* const* d_in, const int* in_sizes, int n_in,
                              void* d_out, int out_size, void* d_ws, size_t ws_size,
                              hipStream_t stream) {
  const float* pred   = (const float*)d_in[0];
  const int*   target = (const int*)d_in[1];
  float*       out    = (float*)d_out;

  char* ws = (char*)d_ws;
  float*    soft   = (float*)ws;
  int*      labels = (int*)(ws + SOFT_BYTES);
  unsigned* cnt    = (unsigned*)(ws + SOFT_BYTES + LABEL_BYTES);
  int*      comp   = (int*)(cnt + 4);

  zero_cnt_kernel<<<1, 256, 0, stream>>>(cnt);
  softmax_kernel<<<BB * HW / 256, 256, 0, stream>>>(pred, soft);            // 4096 blks
  boundary_bce_kernel<<<BC * 64, 256, 0, stream>>>(soft, target, cnt);      // 4864 blks
  cc_init_kernel<<<BIGL / 256, 256, 0, stream>>>(soft, target, labels);     // 155648 blks
  for (int r = 0; r < 6; ++r) {
    cc_prop_kernel<<<NPLANES * 256, 256, 0, stream>>>(labels);              // 38912 blks
    cc_jump_kernel<<<BIGL / 256, 256, 0, stream>>>(labels);
  }
  cc_count_kernel<<<BIGL / 1024, 256, 0, stream>>>(labels, comp);
  finalize_kernel<<<1, 128, 0, stream>>>(cnt, comp, out);
}
